// BiagramLanguageModel_32195074851095
// MI455X (gfx1250) — matrix operand order repaired and checked by reading, not compiled, not run on hardware
//
#include <hip/hip_runtime.h>
#include <hip/hip_bf16.h>
#include <cstdint>

typedef __attribute__((ext_vector_type(16))) _Float16 v16h;
typedef __attribute__((ext_vector_type(8)))  _Float16 v8h;
typedef __attribute__((ext_vector_type(8)))  float    v8f;

#define NEG_BIG (-3.0e38f)

// ---- WMMA fragment maps (CDNA5 ISA 7.12.2, wave32) ----
// A (16x32 f16, MxK): lane L row m=L&15; halves 0..7 at k=((L>>4)<<3), halves 8..15 at +16
// B (32x16 f16, KxN): lane L col n=L&15; halves 0..7 at k=((L>>4)<<3), halves 8..15 at +16 (same as A)
// C/D (16x16 f32): lane L, vgpr j -> m = j + 8*(L>=16), n = L&15
__device__ __forceinline__ v16h cat8(v8h lo, v8h hi) {
    return __builtin_shufflevector(lo, hi, 0, 1, 2, 3, 4, 5, 6, 7,
                                   8, 9, 10, 11, 12, 13, 14, 15);
}

// CDNA5 async DMA: global -> LDS, 16B per lane, tracked by ASYNCcnt.
__device__ __forceinline__ void async_b128(unsigned lds_off, const void* gaddr) {
    asm volatile("global_load_async_to_lds_b128 %0, %1, off"
                 :: "v"(lds_off), "v"((unsigned long long)(uintptr_t)gaddr)
                 : "memory");
}
// async completions retire in order per wave -> threshold waits give double buffering
__device__ __forceinline__ void wait_async0() {
    asm volatile("s_wait_asynccnt 0x0" ::: "memory");
}
__device__ __forceinline__ void wait_async4() {
    asm volatile("s_wait_asynccnt 0x4" ::: "memory");
}
__device__ __forceinline__ unsigned lds_off_of(const void* p) {
    return (unsigned)(uintptr_t)p;   // low 32 bits of generic LDS addr = LDS offset
}

// =====================================================================
// Embedding + loss zeroing.
// =====================================================================
__global__ __launch_bounds__(256)
void embed_kernel(const int* __restrict__ idx, const float* __restrict__ tok,
                  const float* __restrict__ pos, float* __restrict__ x,
                  float* __restrict__ loss0) {
    const int bt = blockIdx.x;
    const int t  = bt & 2047;             // T = 2048
    const int tokid = idx[bt];
    const float* tr = tok + (size_t)tokid * 512;
    const float* pr = pos + (size_t)t * 512;
    float* xr = x + (size_t)bt * 512;
    for (int d = threadIdx.x; d < 512; d += 256) xr[d] = tr[d] + pr[d];
    if (bt == 0 && threadIdx.x == 0) loss0[0] = 0.0f;
}

// =====================================================================
// LayerNorm over D=512, f32 in -> f16 out.
// =====================================================================
__global__ __launch_bounds__(256)
void layernorm_kernel(const float* __restrict__ x, const float* __restrict__ g,
                      const float* __restrict__ b, _Float16* __restrict__ y) {
    __shared__ float s1[256], s2[256];
    const int row = blockIdx.x, tid = threadIdx.x;
    const float* xr = x + (size_t)row * 512;
    const float a0 = xr[tid], a1 = xr[tid + 256];
    s1[tid] = a0 + a1;
    s2[tid] = a0 * a0 + a1 * a1;
    __syncthreads();
    for (int s = 128; s > 0; s >>= 1) {
        if (tid < s) { s1[tid] += s1[tid + s]; s2[tid] += s2[tid + s]; }
        __syncthreads();
    }
    const float mu  = s1[0] * (1.0f / 512.0f);
    const float var = s2[0] * (1.0f / 512.0f) - mu * mu;
    const float inv = rsqrtf(var + 1e-5f);
    _Float16* yr = y + (size_t)row * 512;
    yr[tid]       = (_Float16)((a0 - mu) * inv * g[tid]       + b[tid]);
    yr[tid + 256] = (_Float16)((a1 - mu) * inv * g[tid + 256] + b[tid + 256]);
}

// =====================================================================
// Weight prep: transpose+convert f32[K,N] -> f16[N,K]; and QKV repack.
// =====================================================================
__global__ __launch_bounds__(256)
void transpose_f16_kernel(const float* __restrict__ in, _Float16* __restrict__ out,
                          int K, int N) {
    const int i = blockIdx.x * 256 + threadIdx.x;     // output index, [N][K]
    if (i >= K * N) return;
    const int n = i / K, k = i - n * K;
    out[i] = (_Float16)in[(size_t)k * N + n];
}

__global__ __launch_bounds__(256)
void repack_wt_kernel(const float* __restrict__ in, _Float16* __restrict__ out) {
    // in: (H=8, D=512, HS=64); out: [n=h*64+e][d] f16 (512x512)
    const int i = blockIdx.x * 256 + threadIdx.x;
    const int n = i >> 9, d = i & 511;
    const int h = n >> 6, e = n & 63;
    out[i] = (_Float16)in[(size_t)h * (512 * 64) + (size_t)d * 64 + e];
}

// =====================================================================
// WMMA GEMM: C[M,N] = act(A[M,K] @ Wt[N,K]^T + bias) (+ residual)
// Compile-time flags; async double-buffered LDS staging.
// OUTM: 0 = f32 C, 1 = f16 C, 2 = f16 C transposed [N,M].
// =====================================================================
template <int BIAS, int RELU, int RES, int OUTM>
__global__ __launch_bounds__(128)
void gemm_wmma_kernel(const _Float16* __restrict__ A, const _Float16* __restrict__ Wt,
                      const float* __restrict__ bias, const float* __restrict__ residual,
                      float* __restrict__ C32, _Float16* __restrict__ C16,
                      int M, int N, int K) {
    __shared__ alignas(16) _Float16 sA[2][64 * 40];   // [m][k]
    __shared__ alignas(16) _Float16 sW[2][64 * 40];   // [n][k]

    const int tid  = threadIdx.x;
    const int lane = tid & 31;
    const int wave = tid >> 5;
    const int wm   = (wave >> 1) * 32;
    const int wn   = (wave &  1) * 32;
    const int m0   = blockIdx.y * 64;
    const int n0   = blockIdx.x * 64;

    // staging: 64 rows x 4 chunks of 8 halves, 2 rounds of 128 threads
    const int row0 = tid >> 2;
    const int c8   = (tid & 3) * 8;
    const _Float16* aP0 = A  + (size_t)(m0 + row0) * K + c8;
    const _Float16* aP1 = aP0 + (size_t)32 * K;
    const _Float16* wP0 = Wt + (size_t)(n0 + row0) * K + c8;
    const _Float16* wP1 = wP0 + (size_t)32 * K;
    unsigned la0[2], la1[2], lw0[2], lw1[2];
#pragma unroll
    for (int bf = 0; bf < 2; ++bf) {
        la0[bf] = lds_off_of(&sA[bf][row0 * 40 + c8]);
        la1[bf] = lds_off_of(&sA[bf][(row0 + 32) * 40 + c8]);
        lw0[bf] = lds_off_of(&sW[bf][row0 * 40 + c8]);
        lw1[bf] = lds_off_of(&sW[bf][(row0 + 32) * 40 + c8]);
    }

    v8f acc[2][2] = {};

    auto issue = [&](int bf) {
        async_b128(la0[bf], aP0);
        async_b128(la1[bf], aP1);
        async_b128(lw0[bf], wP0);
        async_b128(lw1[bf], wP1);
        aP0 += 32; aP1 += 32; wP0 += 32; wP1 += 32;
    };
    auto compute = [&](int bf) {
        v16h a[2], b[2];
#pragma unroll
        for (int mt = 0; mt < 2; ++mt) {
            const _Float16* pa = &sA[bf][(wm + mt * 16 + (lane & 15)) * 40
                                         + ((lane >> 4) << 3)];
            a[mt] = cat8(*reinterpret_cast<const v8h*>(pa),
                         *reinterpret_cast<const v8h*>(pa + 16));
        }
#pragma unroll
        for (int nt = 0; nt < 2; ++nt) {
            const _Float16* pw = &sW[bf][(wn + nt * 16 + (lane & 15)) * 40
                                         + ((lane >> 4) << 3)];
            b[nt] = cat8(*reinterpret_cast<const v8h*>(pw),
                         *reinterpret_cast<const v8h*>(pw + 16));
        }
#pragma unroll
        for (int mt = 0; mt < 2; ++mt)
#pragma unroll
            for (int nt = 0; nt < 2; ++nt)
                acc[mt][nt] = __builtin_amdgcn_wmma_f32_16x16x32_f16(
                    false, a[mt], false, b[nt], (short)0, acc[mt][nt], false, false);
    };

    const int ntiles = K >> 5;
    issue(0);                                        // prologue: tile 0
    for (int i = 0; i < ntiles - 1; ++i) {
        issue((i + 1) & 1);                          // tile i+1 in flight
        __builtin_prefetch(aP0, 0, 1);               // ~2 tiles ahead, speculative
        __builtin_prefetch(wP0, 0, 1);
        wait_async4();                               // tile i complete (in-order)
        __syncthreads();
        compute(i & 1);
        __syncthreads();
    }
    wait_async0();
    __syncthreads();
    compute((ntiles - 1) & 1);

    // ---- epilogue (compile-time specialized, batched residual loads)
#pragma unroll
    for (int mt = 0; mt < 2; ++mt)
#pragma unroll
        for (int nt = 0; nt < 2; ++nt) {
            const int mBase = m0 + wm + mt * 16 + ((lane >> 4) << 3);
            const int n = n0 + wn + nt * 16 + (lane & 15);
            const float bv = BIAS ? bias[n] : 0.0f;
            if (OUTM == 2) {                          // packed f16, contiguous in m
                v8h pk;
#pragma unroll
                for (int j = 0; j < 8; ++j) {
                    float v = acc[mt][nt][j] + bv;
                    if (RELU) v = fmaxf(v, 0.0f);
                    pk[j] = (_Float16)v;
                }
                *reinterpret_cast<v8h*>(C16 + (size_t)n * M + mBase) = pk;
            } else {
                const size_t cb = (size_t)mBase * N + n;
                float rv[8];
                if (RES) {
#pragma unroll
                    for (int j = 0; j < 8; ++j) rv[j] = residual[cb + (size_t)j * N];
                }
#pragma unroll
                for (int j = 0; j < 8; ++j) {
                    float v = acc[mt][nt][j] + bv;
                    if (RELU) v = fmaxf(v, 0.0f);
                    if (RES)  v += rv[j];
                    if (OUTM == 0) C32[cb + (size_t)j * N] = v;
                    else           C16[cb + (size_t)j * N] = (_Float16)v;
                }
            }
        }
}

// =====================================================================
// Flash attention with WMMA, async double-buffered K/V staging.
// Q,K: [B*T, 512] f16; Vt: [512, B*T] f16. Block = (b,h,64 q rows), 4 waves.
// =====================================================================
__global__ __launch_bounds__(128)
void attn_kernel(const _Float16* __restrict__ Q, const _Float16* __restrict__ K,
                 const _Float16* __restrict__ Vt, _Float16* __restrict__ O) {
    constexpr int T = 2048;
    constexpr float SCALE = 0.044194173824159216f;   // 1/sqrt(512) per reference

    __shared__ alignas(16) _Float16 sK[2][32 * 72];  // [s][e]
    __shared__ alignas(16) _Float16 sV[2][64 * 40];  // [e][s]
    __shared__ alignas(16) _Float16 sP[4 * 16 * 32];

    const int tid  = threadIdx.x;
    const int lane = tid & 31;
    const int wave = tid >> 5;
    const int qb   = blockIdx.x;
    const int h    = blockIdx.y;
    const int bz   = blockIdx.z;
    const int qt0  = qb * 64 + wave * 16;
    const size_t rowbase = (size_t)bz * T;

    // Q fragments (persist)
    v16h aQ[2];
#pragma unroll
    for (int kh = 0; kh < 2; ++kh) {
        const _Float16* pq = Q + (rowbase + qt0 + (lane & 15)) * 512 + h * 64
                             + kh * 32 + ((lane >> 4) << 3);
        aQ[kh] = cat8(*reinterpret_cast<const v8h*>(pq),
                      *reinterpret_cast<const v8h*>(pq + 16));
    }

    const int krow = tid >> 3, kc8 = (tid & 7) * 8;     // K tile: 32 x 64 halves
    const _Float16* kP0 = K + (rowbase + krow) * 512 + h * 64 + kc8;
    const _Float16* kP1 = kP0 + (size_t)16 * 512;
    const int vrow = tid >> 2, vc8 = (tid & 3) * 8;     // V tile: 64 x 32 halves
    const _Float16* vP0 = Vt + (size_t)(h * 64 + vrow) * 4096 + bz * T + vc8;
    const _Float16* vP1 = vP0 + (size_t)32 * 4096;
    unsigned lk0[2], lk1[2], lv0[2], lv1[2];
#pragma unroll
    for (int bf = 0; bf < 2; ++bf) {
        lk0[bf] = lds_off_of(&sK[bf][krow * 72 + kc8]);
        lk1[bf] = lds_off_of(&sK[bf][(krow + 16) * 72 + kc8]);
        lv0[bf] = lds_off_of(&sV[bf][vrow * 40 + vc8]);
        lv1[bf] = lds_off_of(&sV[bf][(vrow + 32) * 40 + vc8]);
    }

    float mrow[8], lrow[8];
#pragma unroll
    for (int j = 0; j < 8; ++j) { mrow[j] = NEG_BIG; lrow[j] = 0.0f; }
    v8f accO[4] = {};

    auto issue = [&](int bf) {
        async_b128(lk0[bf], kP0);
        async_b128(lk1[bf], kP1);
        async_b128(lv0[bf], vP0);
        async_b128(lv1[bf], vP1);
        kP0 += (size_t)32 * 512; kP1 += (size_t)32 * 512;
        vP0 += 32; vP1 += 32;
    };

    auto compute = [&](int s0, int bf) {
        // ---- S = Q K^T
        v8f sc[2] = {};
#pragma unroll
        for (int st = 0; st < 2; ++st)
#pragma unroll
            for (int kh = 0; kh < 2; ++kh) {
                const _Float16* pk = &sK[bf][(st * 16 + (lane & 15)) * 72
                                            + kh * 32 + ((lane >> 4) << 3)];
                v16h bK = cat8(*reinterpret_cast<const v8h*>(pk),
                               *reinterpret_cast<const v8h*>(pk + 16));
                sc[st] = __builtin_amdgcn_wmma_f32_16x16x32_f16(
                    false, aQ[kh], false, bK, (short)0, sc[st], false, false);
            }

        // ---- scale, causal mask, online softmax
        float fac[8];
#pragma unroll
        for (int j = 0; j < 8; ++j) {
            const int t   = qt0 + j + ((lane >> 4) << 3);
            const int sA_ = s0 + (lane & 15);
            const int sB_ = s0 + 16 + (lane & 15);
            float v0 = sc[0][j] * SCALE; if (sA_ > t) v0 = NEG_BIG;
            float v1 = sc[1][j] * SCALE; if (sB_ > t) v1 = NEG_BIG;
            float tm = fmaxf(v0, v1);
            tm = fmaxf(tm, __shfl_xor(tm, 1));
            tm = fmaxf(tm, __shfl_xor(tm, 2));
            tm = fmaxf(tm, __shfl_xor(tm, 4));
            tm = fmaxf(tm, __shfl_xor(tm, 8));
            const float mnew = fmaxf(mrow[j], tm);
            const float f    = __expf(mrow[j] - mnew);
            mrow[j] = mnew;
            const float p0 = __expf(v0 - mnew);
            const float p1 = __expf(v1 - mnew);
            const int m = j + ((lane >> 4) << 3);
            sP[(wave * 16 + m) * 32 + (lane & 15)]      = (_Float16)p0;
            sP[(wave * 16 + m) * 32 + 16 + (lane & 15)] = (_Float16)p1;
            float rs = p0 + p1;
            rs += __shfl_xor(rs, 1);
            rs += __shfl_xor(rs, 2);
            rs += __shfl_xor(rs, 4);
            rs += __shfl_xor(rs, 8);
            lrow[j] = lrow[j] * f + rs;
            fac[j]  = f;
        }
#pragma unroll
        for (int nt = 0; nt < 4; ++nt)
#pragma unroll
            for (int j = 0; j < 8; ++j) accO[nt][j] *= fac[j];

        // ---- O += P V (wave-private sP; DS ops in order per wave)
        const _Float16* pp = &sP[(wave * 16 + (lane & 15)) * 32 + ((lane >> 4) << 3)];
        v16h aP = cat8(*reinterpret_cast<const v8h*>(pp),
                       *reinterpret_cast<const v8h*>(pp + 16));
#pragma unroll
        for (int nt = 0; nt < 4; ++nt) {
            const _Float16* pv = &sV[bf][(nt * 16 + (lane & 15)) * 40
                                         + ((lane >> 4) << 3)];
            v16h bV = cat8(*reinterpret_cast<const v8h*>(pv),
                           *reinterpret_cast<const v8h*>(pv + 16));
            accO[nt] = __builtin_amdgcn_wmma_f32_16x16x32_f16(
                false, aP, false, bV, (short)0, accO[nt], false, false);
        }
    };

    const int ntiles = qb * 2 + 2;          // s tiles of 32 up to the diagonal
    issue(0);
    for (int i = 0; i < ntiles - 1; ++i) {
        issue((i + 1) & 1);
        wait_async4();
        __syncthreads();
        compute(i * 32, i & 1);
        __syncthreads();
    }
    wait_async0();
    __syncthreads();
    compute((ntiles - 1) * 32, (ntiles - 1) & 1);

    // ---- epilogue: O / l  (f16 out, feeds Wo GEMM)
    const size_t obase = (rowbase + qt0 + ((lane >> 4) << 3)) * 512
                         + h * 64 + (lane & 15);
#pragma unroll
    for (int nt = 0; nt < 4; ++nt)
#pragma unroll
        for (int j = 0; j < 8; ++j)
            O[obase + (size_t)j * 512 + nt * 16] = (_Float16)(accO[nt][j] / lrow[j]);
}

// =====================================================================
// Per-row log-softmax + NLL, atomic-accumulated mean loss.
// =====================================================================
__global__ __launch_bounds__(256)
void loss_kernel(const float* __restrict__ logits, const int* __restrict__ targets,
                 float* __restrict__ loss) {
    __shared__ float red[256];
    const int row = blockIdx.x, tid = threadIdx.x;
    const float* lr = logits + (size_t)row * 32000;

    float mx = NEG_BIG;
    for (int v = tid; v < 32000; v += 256) mx = fmaxf(mx, lr[v]);
    red[tid] = mx; __syncthreads();
    for (int s = 128; s > 0; s >>= 1) {
        if (tid < s) red[tid] = fmaxf(red[tid], red[tid + s]);
        __syncthreads();
    }
    mx = red[0]; __syncthreads();

    float sum = 0.0f;
    for (int v = tid; v < 32000; v += 256) sum += __expf(lr[v] - mx);
    red[tid] = sum; __syncthreads();
    for (int s = 128; s > 0; s >>= 1) {
        if (tid < s) red[tid] += red[tid + s];
        __syncthreads();
    }
    if (tid == 0) {
        const float lse = mx + logf(red[0]);
        const int tgt = targets[row];
        atomicAdd(loss, -(lr[tgt] - lse) * (1.0f / 4096.0f));
    }
}

// =====================================================================
// Host-side orchestration.
// =====================================================================
extern "C" void kernel_launch(void* const* d_in, const int* in_sizes, int n_in,
                              void* d_out, int out_size, void* d_ws, size_t ws_size,
                              hipStream_t stream) {
    (void)in_sizes; (void)n_in; (void)out_size; (void)ws_size;

    const int*   idx     = (const int*)  d_in[0];
    const int*   targets = (const int*)  d_in[1];
    const float* tok_emb = (const float*)d_in[2];
    const float* pos_emb = (const float*)d_in[3];
    const float* ln1_g   = (const float*)d_in[4];
    const float* ln1_b   = (const float*)d_in[5];
    const float* Wq      = (const float*)d_in[6];
    const float* Wk      = (const float*)d_in[7];
    const float* Wv      = (const float*)d_in[8];
    const float* Wo      = (const float*)d_in[9];
    const float* bo      = (const float*)d_in[10];
    const float* ln2_g   = (const float*)d_in[11];
    const float* ln2_b   = (const float*)d_in[12];
    const float* W1      = (const float*)d_in[13];
    const float* b1      = (const float*)d_in[14];
    const float* W2      = (const float*)d_in[15];
    const float* b2      = (const float*)d_in[16];
    const float* lnf_g   = (const float*)d_in[17];
    const float* lnf_b   = (const float*)d_in[18];
    const float* Wout    = (const float*)d_in[19];
    const float* bout    = (const float*)d_in[20];

    float* out = (float*)d_out;            // out[0]=loss, out[1..]=logits
    char*  ws  = (char*)d_ws;
    const size_t MB = 1024u * 1024u;
    float*     xb    = (float*)    (ws + 0 * MB);           // [4096,512] f32
    _Float16*  lnb   = (_Float16*) (ws + 8 * MB);           // [4096,512] f16
    _Float16*  qb    = (_Float16*) (ws + 12 * MB);
    _Float16*  kb    = (_Float16*) (ws + 16 * MB);
    _Float16*  vt    = (_Float16*) (ws + 20 * MB);          // [512,4096] f16 (transposed)
    _Float16*  att   = (_Float16*) (ws + 24 * MB);
    _Float16*  ffh   = (_Float16*) (ws + 28 * MB);          // [4096,2048] f16
    _Float16*  wqt   = (_Float16*) (ws + 44 * MB);
    _Float16*  wkt   = (_Float16*) (ws + 44 * MB + 512 * 1024);
    _Float16*  wvt   = (_Float16*) (ws + 45 * MB);
    _Float16*  wot   = (_Float16*) (ws + 45 * MB + 512 * 1024);
    _Float16*  w1t   = (_Float16*) (ws + 46 * MB);          // [2048,512] f16
    _Float16*  w2t   = (_Float16*) (ws + 48 * MB);          // [512,2048] f16
    _Float16*  woutt = (_Float16*) (ws + 50 * MB);          // [32000,512] f16

    const int BT = 4096;
    const dim3 g512(512 / 64, BT / 64);
    const dim3 gFF(2048 / 64, BT / 64);
    const dim3 gOut(32000 / 64, BT / 64);
    const dim3 gAtt(2048 / 64, 8, 2);

    embed_kernel<<<BT, 256, 0, stream>>>(idx, tok_emb, pos_emb, xb, out);
    transpose_f16_kernel<<<(512 * 32000 + 255) / 256, 256, 0, stream>>>(Wout, woutt, 512, 32000);

    for (int l = 0; l < 4; ++l) {
        const size_t wqk = (size_t)l * 512 * 512;
        layernorm_kernel<<<BT, 256, 0, stream>>>(xb, ln1_g + l * 512, ln1_b + l * 512, lnb);
        repack_wt_kernel<<<1024, 256, 0, stream>>>(Wq + wqk, wqt);
        repack_wt_kernel<<<1024, 256, 0, stream>>>(Wk + wqk, wkt);
        repack_wt_kernel<<<1024, 256, 0, stream>>>(Wv + wqk, wvt);
        transpose_f16_kernel<<<1024, 256, 0, stream>>>(Wo + wqk, wot, 512, 512);
        transpose_f16_kernel<<<4096, 256, 0, stream>>>(W1 + (size_t)l * 512 * 2048, w1t, 512, 2048);
        transpose_f16_kernel<<<4096, 256, 0, stream>>>(W2 + (size_t)l * 2048 * 512, w2t, 2048, 512);

        gemm_wmma_kernel<0, 0, 0, 1><<<g512, 128, 0, stream>>>(
            lnb, wqt, nullptr, nullptr, nullptr, qb, BT, 512, 512);
        gemm_wmma_kernel<0, 0, 0, 1><<<g512, 128, 0, stream>>>(
            lnb, wkt, nullptr, nullptr, nullptr, kb, BT, 512, 512);
        gemm_wmma_kernel<0, 0, 0, 2><<<g512, 128, 0, stream>>>(
            lnb, wvt, nullptr, nullptr, nullptr, vt, BT, 512, 512);
        attn_kernel<<<gAtt, 128, 0, stream>>>(qb, kb, vt, att);
        gemm_wmma_kernel<1, 0, 1, 0><<<g512, 128, 0, stream>>>(
            att, wot, bo + l * 512, xb, xb, nullptr, BT, 512, 512);
        layernorm_kernel<<<BT, 256, 0, stream>>>(xb, ln2_g + l * 512, ln2_b + l * 512, lnb);
        gemm_wmma_kernel<1, 1, 0, 1><<<gFF, 128, 0, stream>>>(
            lnb, w1t, b1 + l * 2048, nullptr, nullptr, ffh, BT, 2048, 512);
        gemm_wmma_kernel<1, 0, 1, 0><<<g512, 128, 0, stream>>>(
            ffh, w2t, b2 + l * 512, xb, xb, nullptr, BT, 512, 2048);
    }

    layernorm_kernel<<<BT, 256, 0, stream>>>(xb, lnf_g, lnf_b, lnb);
    gemm_wmma_kernel<1, 0, 0, 0><<<gOut, 128, 0, stream>>>(
        lnb, woutt, bout, nullptr, out + 1, nullptr, BT, 32000, 512);
    loss_kernel<<<BT, 256, 0, stream>>>(out + 1, targets, out);
}
